// Multi_Vary_Line_Model_v1_39479339384983
// MI455X (gfx1250) — matrix operand order repaired and checked by reading, not compiled, not run on hardware
//
#include <hip/hip_runtime.h>
#include <cstdint>
#include <cstddef>

// ---------------------------------------------------------------------------
// Multi_Vary_Line_Model on MI455X (gfx1250).
// Strategy:
//  * hoist line-invariant attention projection to one wide bf16 WMMA GEMM
//  * single persistent kernel for the sequential 16-line x 16-point recurrence:
//    one WG per 16 batch rows (rows are independent), all state in LDS (~308KB)
//  * all GEMMs via v_wmma_f32_16x16x32_bf16 (f32 accum)
//  * async global->LDS staging (gfx1250 ASYNCcnt path) in the wide GEMM
// ---------------------------------------------------------------------------

typedef __bf16 bf16_t;
typedef __attribute__((ext_vector_type(16))) __bf16 v16bf;
typedef __attribute__((ext_vector_type(8)))  __bf16 v8bf;
typedef __attribute__((ext_vector_type(8)))  float  v8f;

constexpr int  BATCH = 256;
constexpr int  CH    = 512;
constexpr int  PTS   = 64;       // H*W
constexpr int  HSZ   = 512;
constexpr int  NL    = 16;       // n_lines
constexpr int  MP    = 16;       // MAX_POINT
constexpr int  TOK   = BATCH * PTS;   // 16384 tokens

// ---------------- WMMA 16x16 tile helpers (bf16 -> f32) --------------------
// A: 16 x K row-major (lda), this wave's fragment (contiguous k order, same as B):
//   lane<16 : row=lane,    K chunk [k..k+15]
//   lane>=16: row=lane-16, K chunk [k+16..k+31]
// B: weights W stored (N x K) row-major; B[k][n] = W[n][k]:
//   lane column = n0 + (lane&15), contiguous 16 K-values at offset (lane>>4)*16.
__device__ __forceinline__ v8f wmma_acc_16x16(const bf16_t* __restrict__ A, int lda,
                                              const bf16_t* __restrict__ Wn, int ldw,
                                              int K) {
  const int lane = threadIdx.x & 31;
  const int hf   = lane >> 4;
  const int l15  = lane & 15;
  const bf16_t* arow = A  + (size_t)l15 * lda + hf * 16;
  const bf16_t* brow = Wn + (size_t)l15 * ldw + hf * 16;
  v8f acc = {0.f, 0.f, 0.f, 0.f, 0.f, 0.f, 0.f, 0.f};
  for (int k = 0; k < K; k += 32) {
    v8bf alo = *(const v8bf*)(arow + k);
    v8bf ahi = *(const v8bf*)(arow + k + 8);
    v16bf a = __builtin_shufflevector(alo, ahi,
                0,1,2,3,4,5,6,7,8,9,10,11,12,13,14,15);
    v16bf b = *(const v16bf*)(brow + k);
    __builtin_prefetch(brow + k + 512, 0, 0);   // global_prefetch_b8 weight stream
    acc = __builtin_amdgcn_wmma_f32_16x16x32_bf16(
              false, a, false, b, (short)0, acc, false, false);
  }
  return acc;
}

// Workgroup-level GEMM: D[16 x N] (f32, ldd) = A[16 x K] (bf16, lda) * W^T,
// W is (N x K) bf16 row-major. 8 waves stripe the N tiles.
__device__ __forceinline__ void wg_gemm(const bf16_t* A, int lda,
                                        const bf16_t* W, int ldw,
                                        int K, int N, float* D, int ldd) {
  const int wave = threadIdx.x >> 5;
  const int lane = threadIdx.x & 31;
  const int hf = lane >> 4, col = lane & 15;
  for (int n0 = wave * 16; n0 < N; n0 += 8 * 16) {
    v8f acc = wmma_acc_16x16(A, lda, W + (size_t)n0 * ldw, ldw, K);
#pragma unroll
    for (int r = 0; r < 8; r++)
      D[(size_t)(hf * 8 + r) * ldd + n0 + col] = acc[r];
  }
}

// ---------------------- small prep kernels ---------------------------------
__global__ void f32_to_bf16_kernel(const float* __restrict__ s,
                                   bf16_t* __restrict__ d, int n) {
  int i = blockIdx.x * 256 + threadIdx.x;
  if (i < n) d[i] = (bf16_t)s[i];
}

// feat (B,C,P) f32 -> token-major bf16: out[(b*P+p)*C + c]
__global__ void feat_tokens_kernel(const float* __restrict__ feat,
                                   bf16_t* __restrict__ out) {
  size_t i = (size_t)blockIdx.x * 256 + threadIdx.x;   // over TOK*CH
  int c = (int)(i % CH);
  size_t tp = i / CH;
  int p = (int)(tp % PTS);
  size_t b = tp / PTS;
  out[i] = (bf16_t)feat[(b * CH + c) * PTS + p];
}

// avg over P, straight to bf16 (GEMM A operand, ldk=CH)
__global__ void avg_kernel(const float* __restrict__ feat,
                           bf16_t* __restrict__ avgbf) {
  int i = blockIdx.x * 256 + threadIdx.x;              // over BATCH*CH
  const float* f = feat + (size_t)i * PTS;
  float s = 0.f;
#pragma unroll 8
  for (int p = 0; p < PTS; p++) s += f[p];
  avgbf[i] = (bf16_t)(s * (1.0f / PTS));
}

// ---------------- wide GEMM: featProj = feat_bf @ att_w1[:, :C]^T + b1 -----
// M=16384 tokens, N=256, K=512.  A-tile staged to LDS with async loads.
__global__ void __launch_bounds__(256)
featproj_kernel(const bf16_t* __restrict__ featbf,
                const bf16_t* __restrict__ w1,      // (256 x 1024) bf16, feat part cols 0..511
                const float*  __restrict__ b1,
                float* __restrict__ proj) {
  __shared__ bf16_t sA[16 * 512];                     // 16 KB A tile
  const int tid = threadIdx.x;
  const int mt  = blockIdx.x >> 1;                    // 1024 token tiles
  const int nb  = (blockIdx.x & 1) * 128;             // N half

  // async global->LDS staging of the full 16x512 bf16 A tile (4 x b128 / thread)
  const bf16_t* gA = featbf + (size_t)mt * 16 * 512;
  uint32_t lbase = (uint32_t)(uintptr_t)(&sA[0]);
#pragma unroll
  for (int i = 0; i < 4; i++) {
    uint32_t loff = lbase + tid * 16 + i * 4096;
    uint64_t ga   = (uint64_t)(uintptr_t)gA + tid * 16 + i * 4096;
    asm volatile("global_load_async_to_lds_b128 %0, %1, off"
                 :: "v"(loff), "v"(ga) : "memory");
  }
  asm volatile("s_wait_asynccnt 0" ::: "memory");
  __syncthreads();

  const int wave = tid >> 5;
  const int n0 = nb + wave * 16;
  v8f acc = wmma_acc_16x16(sA, 512, w1 + (size_t)n0 * 1024, 1024, 512);
  const int lane = tid & 31, hf = lane >> 4, col = lane & 15;
#pragma unroll
  for (int r = 0; r < 8; r++) {
    int row = hf * 8 + r;
    proj[((size_t)mt * 16 + row) * 256 + n0 + col] = acc[r] + b1[n0 + col];
  }
}

// --------------------- persistent recurrent kernel -------------------------
struct PParams {
  const float*  feat;       // (B,C,P) f32
  const float*  featProj;   // (TOK,256) f32 (includes att_b1)
  const bf16_t* avgbf;      // (B,C) bf16
  const bf16_t* hp_w1;  const float* hp_b1;
  const bf16_t* hp_w2;  const float* hp_b2;
  const bf16_t* att_w1;                      // (256 x 1024) bf16
  const float*  att_w2; const float* att_b2;
  const bf16_t* sg_wih; const bf16_t* sg_whh;
  const float*  sg_bih; const float*  sg_bhh;
  const bf16_t* pg_wih; const bf16_t* pg_whh;
  const float*  pg_bih; const float*  pg_bhh;
  const float*  pp_w;   const float* pp_b;
  const float*  lp_w;   const float* lp_b;
  float* out;
};

// LDS layout (dynamic): fits CDNA5's 320KB/WGP budget
constexpr int OFF_GI  = 0;                       // 16x1536 f32  (96KB)
constexpr int OFF_GH  = OFF_GI  + 16*1536*4;     // 16x1536 f32  (96KB)
constexpr int OFF_H   = OFF_GH  + 16*1536*4;     // 16x512 f32 sh (32KB)
constexpr int OFF_PH  = OFF_H   + 16*512*4;      // 16x512 f32 ph (32KB)
constexpr int OFF_X   = OFF_PH  + 16*512*4;      // 16x512 f32 scratch (32KB)
constexpr int OFF_AB  = OFF_X   + 16*512*4;      // 16x512 bf16 A staging (16KB)
constexpr int OFF_ATT = OFF_AB  + 16*512*2;      // 16x64 f32 att scores (4KB)
constexpr int OFF_SM  = OFF_ATT + 16*64*4;       // 16x4 f32 tiny dots
constexpr int SMEM_BYTES = OFF_SM + 16*4*4;      // 315,648 B < 327,680 B

__device__ __forceinline__ void conv_to_bf(const float* __restrict__ S,
                                           bf16_t* __restrict__ D, int n) {
  for (int i = threadIdx.x; i < n; i += 256) D[i] = (bf16_t)S[i];
}

__device__ __forceinline__ void gru_elem(const float* __restrict__ GI,
                                         const float* __restrict__ GH,
                                         const float* __restrict__ bih,
                                         const float* __restrict__ bhh,
                                         float* __restrict__ H) {
  for (int i = threadIdx.x; i < 16 * 512; i += 256) {
    int r = i >> 9, j = i & 511;
    float ir = GI[r*1536 + j]        + bih[j]        + GH[r*1536 + j]        + bhh[j];
    float iz = GI[r*1536 + 512 + j]  + bih[512 + j]  + GH[r*1536 + 512 + j]  + bhh[512 + j];
    float in_ = GI[r*1536 + 1024 + j] + bih[1024 + j];
    float hn  = GH[r*1536 + 1024 + j] + bhh[1024 + j];
    float rg = 1.f / (1.f + __expf(-ir));
    float zg = 1.f / (1.f + __expf(-iz));
    float ng = tanhf(in_ + rg * hn);
    H[r*512 + j] = (1.f - zg) * ng + zg * H[r*512 + j];
  }
}

__global__ void __launch_bounds__(256)
persistent_kernel(PParams p) {
  extern __shared__ char smem[];
  float*  sGI = (float*)(smem + OFF_GI);
  float*  sGH = (float*)(smem + OFF_GH);
  float*  sH  = (float*)(smem + OFF_H);
  float*  sPH = (float*)(smem + OFF_PH);
  float*  sX  = (float*)(smem + OFF_X);
  bf16_t* sAb = (bf16_t*)(smem + OFF_AB);
  float*  sAt = (float*)(smem + OFF_ATT);
  float*  sSm = (float*)(smem + OFF_SM);

  const int wg = blockIdx.x;                 // 16 WGs, 16 batch rows each
  const int tid = threadIdx.x;

  // ---- h0 = relu(avg @ hp_w1^T + b1) @ hp_w2^T + b2 ----
  wg_gemm(p.avgbf + (size_t)wg * 16 * CH, CH, p.hp_w1, CH, CH, 256, sX, 256);
  __syncthreads();
  for (int i = tid; i < 16 * 256; i += 256) {
    float v = sX[i] + p.hp_b1[i & 255];
    sAb[i] = (bf16_t)fmaxf(v, 0.f);
  }
  __syncthreads();
  wg_gemm(sAb, 256, p.hp_w2, 256, 256, HSZ, sH, HSZ);
  __syncthreads();
  for (int i = tid; i < 16 * HSZ; i += 256) sH[i] += p.hp_b2[i & 511];
  __syncthreads();

  for (int l = 0; l < NL; l++) {
    // ---- state projection: sh @ att_w1[:, C:]^T -> sX (16x256) ----
    conv_to_bf(sH, sAb, 16 * HSZ);
    __syncthreads();
    wg_gemm(sAb, HSZ, p.att_w1 + 512, 1024, HSZ, 256, sX, 256);
    __syncthreads();

    // ---- attention scores per token ----
    for (int job = tid; job < 16 * PTS; job += 256) {
      int row = job >> 6, pp = job & 63;
      size_t token = ((size_t)(wg * 16 + row)) * PTS + pp;
      const float* fp = p.featProj + token * 256;
      const float* sp = sX + row * 256;
      float s = 0.f;
#pragma unroll 4
      for (int o = 0; o < 256; o++)
        s += fmaxf(fp[o] + sp[o], 0.f) * p.att_w2[o];
      sAt[job] = s + p.att_b2[0];
    }
    __syncthreads();

    // ---- att_feat[b][c] = sum_p feat[b][c][p] * att[b][p] -> sX (16x512) ----
    for (int i = tid; i < 16 * CH; i += 256) {
      int row = i >> 9, c = i & 511;
      const float* f = p.feat + (((size_t)(wg * 16 + row)) * CH + c) * PTS;
      const float* a = sAt + row * PTS;
      float s = 0.f;
#pragma unroll 8
      for (int pp = 0; pp < PTS; pp++) s += f[pp] * a[pp];
      sX[i] = s;
    }
    __syncthreads();

    // ---- line GRU ----
    conv_to_bf(sX, sAb, 16 * CH);
    __syncthreads();
    wg_gemm(sAb, CH, p.sg_wih, CH, CH, 3 * HSZ, sGI, 3 * HSZ);
    __syncthreads();
    conv_to_bf(sH, sAb, 16 * HSZ);
    __syncthreads();
    wg_gemm(sAb, HSZ, p.sg_whh, HSZ, HSZ, 3 * HSZ, sGH, 3 * HSZ);
    __syncthreads();
    gru_elem(sGI, sGH, p.sg_bih, p.sg_bhh, sH);   // sH <- sh_new
    __syncthreads();

    // ---- line_p = relu(sh_new) @ lp_w^T + lp_b ; softmax -> out ----
    if (tid < 32) {
      int row = tid >> 1, k = tid & 1;
      const float* w = p.lp_w + k * HSZ;
      float s = p.lp_b[k];
      for (int j = 0; j < HSZ; j++) s += fmaxf(sH[row * HSZ + j], 0.f) * w[j];
      sSm[row * 2 + k] = s;
    }
    __syncthreads();
    if (tid < 16) {
      int row = tid, bidx = wg * 16 + row;
      float a = sSm[row * 2], b = sSm[row * 2 + 1];
      float m = fmaxf(a, b);
      float e0 = __expf(a - m), e1 = __expf(b - m), inv = 1.f / (e0 + e1);
      size_t base = 262144 + ((size_t)bidx * NL + l) * 2;
      p.out[base + 0] = e0 * inv;
      p.out[base + 1] = e1 * inv;
    }

    // ---- point loop: gi is line-invariant (x = sh_new constant) ----
    for (int i = tid; i < 16 * HSZ; i += 256) sPH[i] = 0.f;
    conv_to_bf(sH, sAb, 16 * HSZ);
    __syncthreads();
    wg_gemm(sAb, HSZ, p.pg_wih, HSZ, HSZ, 3 * HSZ, sGI, 3 * HSZ);
    __syncthreads();

    for (int t = 0; t < MP; t++) {
      conv_to_bf(sPH, sAb, 16 * HSZ);
      __syncthreads();
      wg_gemm(sAb, HSZ, p.pg_whh, HSZ, HSZ, 3 * HSZ, sGH, 3 * HSZ);
      __syncthreads();
      gru_elem(sGI, sGH, p.pg_bih, p.pg_bhh, sPH);
      __syncthreads();
      if (tid < 64) {
        int row = tid >> 2, k = tid & 3;
        const float* w = p.pp_w + k * HSZ;
        float s = p.pp_b[k];
        for (int j = 0; j < HSZ; j++) s += fmaxf(sPH[row * HSZ + j], 0.f) * w[j];
        sSm[row * 4 + k] = s;
      }
      __syncthreads();
      if (tid < 16) {
        int row = tid, bidx = wg * 16 + row;
        size_t base = (((size_t)bidx * NL + l) * MP + t) * 2;
        p.out[base + 0] = sSm[row * 4 + 0];
        p.out[base + 1] = sSm[row * 4 + 1];
        float a = sSm[row * 4 + 2], b = sSm[row * 4 + 3];
        float m = fmaxf(a, b);
        float e0 = __expf(a - m), e1 = __expf(b - m), inv = 1.f / (e0 + e1);
        p.out[131072 + base + 0] = e0 * inv;
        p.out[131072 + base + 1] = e1 * inv;
      }
      __syncthreads();
    }
  }
}

// ----------------------------- launcher ------------------------------------
extern "C" void kernel_launch(void* const* d_in, const int* in_sizes, int n_in,
                              void* d_out, int out_size, void* d_ws, size_t ws_size,
                              hipStream_t stream) {
  const float* feat    = (const float*)d_in[0];
  const float* hp_w1   = (const float*)d_in[1];
  const float* hp_b1   = (const float*)d_in[2];
  const float* hp_w2   = (const float*)d_in[3];
  const float* hp_b2   = (const float*)d_in[4];
  const float* att_w1  = (const float*)d_in[5];
  const float* att_b1  = (const float*)d_in[6];
  const float* att_w2  = (const float*)d_in[7];
  const float* att_b2  = (const float*)d_in[8];
  const float* sg_wih  = (const float*)d_in[9];
  const float* sg_whh  = (const float*)d_in[10];
  const float* sg_bih  = (const float*)d_in[11];
  const float* sg_bhh  = (const float*)d_in[12];
  const float* pg_wih  = (const float*)d_in[13];
  const float* pg_whh  = (const float*)d_in[14];
  const float* pg_bih  = (const float*)d_in[15];
  const float* pg_bhh  = (const float*)d_in[16];
  const float* pp_w    = (const float*)d_in[17];
  const float* pp_b    = (const float*)d_in[18];
  const float* lp_w    = (const float*)d_in[19];
  const float* lp_b    = (const float*)d_in[20];
  // d_in[21] = n_lines (16; output layout is fixed by harness)

  char* ws = (char*)d_ws;
  size_t cur = 0;
  auto take = [&](size_t bytes) -> char* {
    char* r = ws + cur;
    cur += (bytes + 255) & ~(size_t)255;
    return r;
  };
  bf16_t* feat_bf  = (bf16_t*)take((size_t)TOK * CH * 2);
  float*  featProj = (float*) take((size_t)TOK * 256 * 4);
  bf16_t* avg_bf   = (bf16_t*)take((size_t)BATCH * CH * 2);
  bf16_t* hp_w1b   = (bf16_t*)take((size_t)256 * CH * 2);
  bf16_t* hp_w2b   = (bf16_t*)take((size_t)HSZ * 256 * 2);
  bf16_t* att_w1b  = (bf16_t*)take((size_t)256 * 1024 * 2);
  bf16_t* sg_wihb  = (bf16_t*)take((size_t)3 * HSZ * CH * 2);
  bf16_t* sg_whhb  = (bf16_t*)take((size_t)3 * HSZ * HSZ * 2);
  bf16_t* pg_wihb  = (bf16_t*)take((size_t)3 * HSZ * HSZ * 2);
  bf16_t* pg_whhb  = (bf16_t*)take((size_t)3 * HSZ * HSZ * 2);

  auto cvt = [&](const float* s, bf16_t* d, int n) {
    f32_to_bf16_kernel<<<(n + 255) / 256, 256, 0, stream>>>(s, d, n);
  };
  cvt(hp_w1,  hp_w1b,  256 * CH);
  cvt(hp_w2,  hp_w2b,  HSZ * 256);
  cvt(att_w1, att_w1b, 256 * 1024);
  cvt(sg_wih, sg_wihb, 3 * HSZ * CH);
  cvt(sg_whh, sg_whhb, 3 * HSZ * HSZ);
  cvt(pg_wih, pg_wihb, 3 * HSZ * HSZ);
  cvt(pg_whh, pg_whhb, 3 * HSZ * HSZ);

  feat_tokens_kernel<<<(TOK * CH) / 256, 256, 0, stream>>>(feat, feat_bf);
  avg_kernel<<<(BATCH * CH) / 256, 256, 0, stream>>>(feat, avg_bf);
  featproj_kernel<<<2048, 256, 0, stream>>>(feat_bf, att_w1b, att_b1, featProj);

  PParams pp;
  pp.feat = feat;       pp.featProj = featProj; pp.avgbf = avg_bf;
  pp.hp_w1 = hp_w1b;    pp.hp_b1 = hp_b1;
  pp.hp_w2 = hp_w2b;    pp.hp_b2 = hp_b2;
  pp.att_w1 = att_w1b;  pp.att_w2 = att_w2;     pp.att_b2 = att_b2;
  pp.sg_wih = sg_wihb;  pp.sg_whh = sg_whhb;
  pp.sg_bih = sg_bih;   pp.sg_bhh = sg_bhh;
  pp.pg_wih = pg_wihb;  pp.pg_whh = pg_whhb;
  pp.pg_bih = pg_bih;   pp.pg_bhh = pg_bhh;
  pp.pp_w = pp_w;       pp.pp_b = pp_b;
  pp.lp_w = lp_w;       pp.lp_b = lp_b;
  pp.out = (float*)d_out;

  persistent_kernel<<<BATCH / 16, 256, SMEM_BYTES, stream>>>(pp);
}
